// GIC_Decoder_Core_58334245814772
// MI455X (gfx1250) — matrix operand order repaired and checked by reading, not compiled, not run on hardware
//
#include <hip/hip_runtime.h>
#include <hip/hip_bf16.h>

typedef __attribute__((ext_vector_type(16))) __bf16 v16bf;
typedef __attribute__((ext_vector_type(8)))  float  v8f;

#define D_  512
#define H_  8
#define L_  579
#define B_  256

// d_out layout (floats): output | new_h(2) | new_c(2) | new_pre(2)
#define OFF_OUT    0
#define OFF_NEWH0  131072
#define OFF_NEWH1  262144
#define OFF_NEWC0  393216
#define OFF_NEWC1  524288
#define OFF_PRE0   655360
#define OFF_PRE1   803584   // 655360 + 256*579

__device__ __forceinline__ float sigm(float x) { return 1.0f / (1.0f + __expf(-x)); }

__device__ __forceinline__ float wave_sum(float v) {
  #pragma unroll
  for (int o = 16; o > 0; o >>= 1) v += __shfl_xor(v, o, 32);
  return v;
}
__device__ __forceinline__ float wave_max(float v) {
  #pragma unroll
  for (int o = 16; o > 0; o >>= 1) v = fmaxf(v, __shfl_xor(v, o, 32));
  return v;
}

// ---------------- weight fp32 -> bf16 (optionally strided dest for concat) ----
__global__ void k_convert_bf16(const float* __restrict__ src, __bf16* __restrict__ dst,
                               int src_cols, int dst_stride, int dst_off, int total) {
  int i = blockIdx.x * blockDim.x + threadIdx.x;
  if (i >= total) return;
  int r = i / src_cols, c = i - r * src_cols;
  dst[(size_t)r * dst_stride + dst_off + c] = (__bf16)src[i];
}

// ---------------- build x_in_cat = [xt, mean_vit + h1, h0]  (B x 1536) --------
__global__ void k_build_xin(const float* __restrict__ xt, const float* __restrict__ mean_vit,
                            const float* __restrict__ state_h, float* __restrict__ x_in) {
  int i = blockIdx.x * blockDim.x + threadIdx.x;
  if (i >= B_ * D_) return;
  int b = i >> 9, j = i & 511;
  x_in[(size_t)b * 1536 + j]        = xt[i];
  x_in[(size_t)b * 1536 + 512 + j]  = mean_vit[i] + state_h[B_ * D_ + i]; // h1
  x_in[(size_t)b * 1536 + 1024 + j] = state_h[i];                        // h0
}

// ---------------- generic bf16 WMMA GEMM: C = act(A @ W^T + b1 + b2) ----------
// A: M x K fp32 (lda), W: N x K bf16 row-major (K contiguous), C: M x N fp32 (ldc)
// block = 128 (4 waves, one 16x16 tile each); grid = (N/16, M/64)
__global__ void k_gemm_bf16(const float* __restrict__ A, int lda,
                            const __bf16* __restrict__ W,
                            const float* __restrict__ bias1,
                            const float* __restrict__ bias2,
                            float* __restrict__ C, int ldc, int K, int act) {
  int wave = threadIdx.x >> 5;
  int lane = threadIdx.x & 31;
  int mt = blockIdx.y * 4 + wave;
  int nt = blockIdx.x;
  int mr = lane & 15, hi = lane >> 4;
  int n = nt * 16 + mr;
  const float*  Arow = A + (size_t)(mt * 16 + mr) * lda;
  const __bf16* Wrow = W + (size_t)n * K;
  v8f acc = {};
  for (int k0 = 0; k0 < K; k0 += 32) {
    v16bf a;
    #pragma unroll
    for (int j = 0; j < 8; ++j) {               // A layout: K contiguous, +16 for hi lanes (matches B)
      a[j]     = (__bf16)Arow[k0 + hi * 16 + j];
      a[j + 8] = (__bf16)Arow[k0 + hi * 16 + 8 + j];
    }
    v16bf bfr = *(const v16bf*)(Wrow + k0 + hi * 16); // B layout: K contiguous, +16 for hi lanes (same k order as A)
    acc = __builtin_amdgcn_wmma_f32_16x16x32_bf16(false, a, false, bfr, (short)0, acc, false, false);
  }
  float b1 = bias1 ? bias1[n] : 0.0f;
  float b2 = bias2 ? bias2[n] : 0.0f;
  #pragma unroll
  for (int r = 0; r < 8; ++r) {
    int row = mt * 16 + r + hi * 8;             // ISA C/D layout
    float v = acc[r] + b1 + b2;
    if (act == 1) v = fmaxf(v, 0.0f);
    C[(size_t)row * ldc + n] = v;
  }
}

// ---------------- LSTM cell + h_attn + LayerNorm(q) ---------------------------
// one wave per batch row; grid = 32, block = 256
__global__ void k_lstm_ln(const float* __restrict__ gates, const float* __restrict__ state_c,
                          const float* __restrict__ pre_att,
                          const float* __restrict__ ln_g, const float* __restrict__ ln_b,
                          float* __restrict__ dout, float* __restrict__ ctx_in,
                          float* __restrict__ xq) {
  int wave = threadIdx.x >> 5, lane = threadIdx.x & 31;
  int b = blockIdx.x * 8 + wave;
  const float* g = gates + (size_t)b * 2048;
  float hvals[16];
  float msum = 0.0f;
  #pragma unroll 4
  for (int it = 0; it < 16; ++it) {
    int j = lane + it * 32;
    float ig = g[j], fg = g[512 + j], gg = g[1024 + j], og = g[1536 + j];
    float c0 = state_c[(size_t)b * 512 + j];
    float c_att = sigm(fg) * c0 + sigm(ig) * tanhf(gg);
    float h_att = sigm(og) * tanhf(c_att);
    float h_attn = h_att + pre_att[(size_t)b * 512 + j];
    hvals[it] = h_attn;
    msum += h_attn;
    dout[OFF_NEWC0 + b * 512 + j] = c_att;
    dout[OFF_NEWC1 + b * 512 + j] = state_c[(size_t)B_ * 512 + b * 512 + j];
    dout[OFF_NEWH0 + b * 512 + j] = h_att;
    ctx_in[(size_t)b * 1024 + 512 + j] = h_att;   // ctx concat second half
  }
  float mean = wave_sum(msum) * (1.0f / 512.0f);
  float vsum = 0.0f;
  #pragma unroll
  for (int it = 0; it < 16; ++it) { float d = hvals[it] - mean; vsum += d * d; }
  float rstd = rsqrtf(wave_sum(vsum) * (1.0f / 512.0f) + 1e-5f);
  #pragma unroll
  for (int it = 0; it < 16; ++it) {
    int j = lane + it * 32;
    xq[(size_t)b * 1024 + 512 + j] = ln_g[j] * (hvals[it] - mean) * rstd + ln_b[j];
  }
}

// ---------------- fused single-query attention (one block per batch) ----------
// scores = reV @ (Wk^T qh)  via WMMA; softmax; t = attn @ reV; x = Wv t + bv
__global__ void k_attention(const float* __restrict__ reV, const float* __restrict__ Wfc1,
                            const float* __restrict__ bfc1, const float* __restrict__ qh,
                            const float* __restrict__ pre_attn, float* __restrict__ dout,
                            float* __restrict__ xq) {
  __shared__ float qh_s[512];
  __shared__ float qk_s[8][776];   // stage0: folded query; stage3: reused as t[h][k]
  __shared__ float s_s[8][584];    // scores -> attn
  int b = blockIdx.x;
  int tid = threadIdx.x;
  int wave = tid >> 5, lane = tid & 31;

  qh_s[tid]       = qh[(size_t)b * 512 + tid];
  qh_s[tid + 256] = qh[(size_t)b * 512 + tid + 256];
  __syncthreads();

  { // stage 0: qk[h][k] = sum_d qh[h*64+d] * Wkey[h*64+d][k]   (wave w -> head w)
    int h = wave;
    for (int c = 0; c < 24; ++c) {
      int k = c * 32 + lane;
      const float* wk = Wfc1 + (size_t)(512 + h * 64) * 768 + k;
      float acc = 0.0f;
      #pragma unroll 8
      for (int d = 0; d < 64; ++d) acc += qh_s[h * 64 + d] * wk[(size_t)d * 768];
      qk_s[h][k] = acc;
    }
  }
  __syncthreads();

  { // stage 1: scores via WMMA, M = L-tile(16), N = 16 (8 heads + pad), K = 768
    int mr = lane & 15, hi = lane >> 4;
    int h = mr & 7;
    for (int lt = wave; lt < 37; lt += 8) {
      int row = lt * 16 + mr;
      const float* Arow = reV + ((size_t)b * L_ + row) * 768;
      bool ok = row < L_;
      v8f acc = {};
      for (int kt = 0; kt < 24; ++kt) {
        int k0 = kt * 32;
        v16bf a, bfr;
        if (ok) {
          #pragma unroll
          for (int j = 0; j < 8; ++j) {
            a[j]     = (__bf16)Arow[k0 + hi * 16 + j];
            a[j + 8] = (__bf16)Arow[k0 + hi * 16 + 8 + j];
          }
        } else {
          #pragma unroll
          for (int j = 0; j < 16; ++j) a[j] = (__bf16)0.0f;
        }
        #pragma unroll
        for (int t = 0; t < 16; ++t) bfr[t] = (__bf16)qk_s[h][k0 + hi * 16 + t];
        acc = __builtin_amdgcn_wmma_f32_16x16x32_bf16(false, a, false, bfr, (short)0, acc, false, false);
      }
      if (mr < 8) {
        #pragma unroll
        for (int r = 0; r < 8; ++r) {
          int lr = lt * 16 + r + hi * 8;
          if (lr < L_) s_s[mr][lr] = acc[r] * 0.125f;   // 1/sqrt(64)
        }
      }
    }
  }
  __syncthreads();

  { // stage 2: softmax over L per head (key bias is softmax-invariant -> dropped)
    int h = wave;
    float m = -1e30f;
    for (int l = lane; l < L_; l += 32) m = fmaxf(m, s_s[h][l]);
    m = wave_max(m);
    float sum = 0.0f;
    for (int l = lane; l < L_; l += 32) { float e = __expf(s_s[h][l] - m); s_s[h][l] = e; sum += e; }
    float inv = 1.0f / wave_sum(sum);
    for (int l = lane; l < L_; l += 32) s_s[h][l] *= inv;
  }
  __syncthreads();

  // weight_att = sum_h attn ; copy pre_attn[0]
  for (int l = tid; l < L_; l += 256) {
    float s = 0.0f;
    #pragma unroll
    for (int h = 0; h < 8; ++h) s += s_s[h][l];
    dout[OFF_PRE0 + b * L_ + l] = s;
    dout[OFF_PRE1 + b * L_ + l] = pre_attn[(size_t)b * L_ + l];
  }

  { // stage 3: t[h][k] = sum_l attn[h][l] * reV[b,l,k]  (coalesced; reuse qk_s)
    float acc[3][8];
    #pragma unroll
    for (int c = 0; c < 3; ++c)
      #pragma unroll
      for (int h = 0; h < 8; ++h) acc[c][h] = 0.0f;
    int kbase = wave * 32 + lane;
    for (int l = 0; l < L_; ++l) {
      float p[8];
      #pragma unroll
      for (int h = 0; h < 8; ++h) p[h] = s_s[h][l];
      const float* rv = reV + ((size_t)b * L_ + l) * 768;
      #pragma unroll
      for (int c = 0; c < 3; ++c) {
        float a = rv[kbase + c * 256];
        #pragma unroll
        for (int h = 0; h < 8; ++h) acc[c][h] += p[h] * a;
      }
    }
    __syncthreads();   // qk_s fully consumed by stage 1
    #pragma unroll
    for (int c = 0; c < 3; ++c)
      #pragma unroll
      for (int h = 0; h < 8; ++h) qk_s[h][kbase + c * 256] = acc[c][h];
  }
  __syncthreads();

  { // stage 4: x[h*64+d] = t[h] . Wval[h*64+d] + b_fc1[h*64+d]  (sum(attn)=1)
    int h = wave;
    for (int dd = 0; dd < 64; ++dd) {
      int row = h * 64 + dd;
      const float* wv = Wfc1 + (size_t)row * 768;
      float acc = 0.0f;
      #pragma unroll 4
      for (int j = 0; j < 24; ++j) { int k = j * 32 + lane; acc += qk_s[h][k] * wv[k]; }
      acc = wave_sum(acc);
      if (lane == 0) xq[(size_t)b * 1024 + row] = acc + bfc1[row];
    }
  }
}

// ---------------- GLU + LayerNorm(ff input) -----------------------------------
__global__ void k_glu_ln(const float* __restrict__ xg, const float* __restrict__ g,
                         const float* __restrict__ be, float* __restrict__ glu_x,
                         float* __restrict__ ln_x) {
  int wave = threadIdx.x >> 5, lane = threadIdx.x & 31;
  int b = blockIdx.x * 8 + wave;
  float xv[16];
  float msum = 0.0f;
  #pragma unroll 4
  for (int it = 0; it < 16; ++it) {
    int j = lane + it * 32;
    float a = xg[(size_t)b * 1024 + j];
    float bg = xg[(size_t)b * 1024 + 512 + j];
    float x = a * sigm(bg);
    xv[it] = x; msum += x;
    glu_x[(size_t)b * 512 + j] = x;
  }
  float mean = wave_sum(msum) * (1.0f / 512.0f);
  float vsum = 0.0f;
  #pragma unroll
  for (int it = 0; it < 16; ++it) { float d = xv[it] - mean; vsum += d * d; }
  float rstd = rsqrtf(wave_sum(vsum) * (1.0f / 512.0f) + 1e-5f);
  #pragma unroll
  for (int it = 0; it < 16; ++it) {
    int j = lane + it * 32;
    ln_x[(size_t)b * 512 + j] = g[j] * (xv[it] - mean) * rstd + be[j];
  }
}

// ---------------- att_out = glu_x + ff  -> ctx_in[:, :512] --------------------
__global__ void k_addto_ctx(const float* __restrict__ glu_x, const float* __restrict__ ff,
                            float* __restrict__ ctx_in) {
  int i = blockIdx.x * blockDim.x + threadIdx.x;
  if (i >= B_ * 512) return;
  int b = i >> 9, j = i & 511;
  ctx_in[(size_t)b * 1024 + j] = glu_x[i] + ff[i];
}

// ---------------- final GLU -> output (and new_h[1]) --------------------------
__global__ void k_final_glu(const float* __restrict__ ctxg, float* __restrict__ dout) {
  int i = blockIdx.x * blockDim.x + threadIdx.x;
  if (i >= B_ * 512) return;
  int b = i >> 9, j = i & 511;
  float a = ctxg[(size_t)b * 1024 + j];
  float g = ctxg[(size_t)b * 1024 + 512 + j];
  float o = a * sigm(g);
  dout[OFF_OUT + i]   = o;
  dout[OFF_NEWH1 + i] = o;
}

extern "C" void kernel_launch(void* const* d_in, const int* in_sizes, int n_in,
                              void* d_out, int out_size, void* d_ws, size_t ws_size,
                              hipStream_t stream) {
  const float* xt       = (const float*)d_in[0];
  const float* re_VIT   = (const float*)d_in[5];
  const float* mean_vit = (const float*)d_in[6];
  const float* pre_attn = (const float*)d_in[7];
  const float* state_h  = (const float*)d_in[8];
  const float* state_c  = (const float*)d_in[9];
  const float* W_fc  = (const float*)d_in[10]; const float* b_fc  = (const float*)d_in[11];
  const float* W_fc1 = (const float*)d_in[12]; const float* b_fc1 = (const float*)d_in[13];
  const float* W_ih  = (const float*)d_in[14]; const float* W_hh  = (const float*)d_in[15];
  const float* b_ih  = (const float*)d_in[16]; const float* b_hh  = (const float*)d_in[17];
  const float* ln_q_g = (const float*)d_in[18]; const float* ln_q_b = (const float*)d_in[19];
  const float* W_q   = (const float*)d_in[20]; const float* b_q   = (const float*)d_in[21];
  const float* W_glu = (const float*)d_in[22]; const float* b_glu = (const float*)d_in[23];
  const float* ln_ff_g = (const float*)d_in[24]; const float* ln_ff_b = (const float*)d_in[25];
  const float* W_ff1 = (const float*)d_in[26]; const float* b_ff1 = (const float*)d_in[27];
  const float* W_ff2 = (const float*)d_in[28]; const float* b_ff2 = (const float*)d_in[29];
  const float* W_ctx = (const float*)d_in[30]; const float* b_ctx = (const float*)d_in[31];
  float* out = (float*)d_out;
  char* ws = (char*)d_ws;
  size_t off = 0;
  auto alloc = [&](size_t bytes) -> char* {
    char* p = ws + off;
    off = (off + bytes + 255) & ~(size_t)255;
    return p;
  };
  // bf16 weights
  __bf16* Wfc_bf  = (__bf16*)alloc((size_t)512 * 576 * 2);
  __bf16* Wcat_bf = (__bf16*)alloc((size_t)2048 * 1536 * 2);  // [W_ih | W_hh]
  __bf16* Wq_bf   = (__bf16*)alloc((size_t)512 * 512 * 2);
  __bf16* Wglu_bf = (__bf16*)alloc((size_t)1024 * 1024 * 2);
  __bf16* Wff1_bf = (__bf16*)alloc((size_t)2048 * 512 * 2);
  __bf16* Wff2_bf = (__bf16*)alloc((size_t)512 * 2048 * 2);
  __bf16* Wctx_bf = (__bf16*)alloc((size_t)1024 * 1024 * 2);
  // fp32 intermediates
  float* x_in   = (float*)alloc((size_t)B_ * 1536 * 4);
  float* preA   = (float*)alloc((size_t)B_ * 512 * 4);
  float* gates  = (float*)alloc((size_t)B_ * 2048 * 4);
  float* qh     = (float*)alloc((size_t)B_ * 512 * 4);
  float* xq     = (float*)alloc((size_t)B_ * 1024 * 4);  // [x_att | q]
  float* xg     = (float*)alloc((size_t)B_ * 1024 * 4);
  float* glu_x  = (float*)alloc((size_t)B_ * 512 * 4);
  float* ln_x   = (float*)alloc((size_t)B_ * 512 * 4);
  float* h_ff   = (float*)alloc((size_t)B_ * 2048 * 4);
  float* ffb    = (float*)alloc((size_t)B_ * 512 * 4);
  float* ctx_in = (float*)alloc((size_t)B_ * 1024 * 4);
  float* ctxg   = (float*)alloc((size_t)B_ * 1024 * 4);
  (void)in_sizes; (void)n_in; (void)out_size; (void)ws_size;

  auto cvt = [&](const float* src, __bf16* dst, int rows, int cols, int dstride, int doff) {
    int total = rows * cols;
    k_convert_bf16<<<(total + 255) / 256, 256, 0, stream>>>(src, dst, cols, dstride, doff, total);
  };
  cvt(W_fc,  Wfc_bf,  512, 576,  576,  0);
  cvt(W_ih,  Wcat_bf, 2048, 1024, 1536, 0);
  cvt(W_hh,  Wcat_bf, 2048, 512,  1536, 1024);
  cvt(W_q,   Wq_bf,   512, 512,  512,  0);
  cvt(W_glu, Wglu_bf, 1024, 1024, 1024, 0);
  cvt(W_ff1, Wff1_bf, 2048, 512,  512,  0);
  cvt(W_ff2, Wff2_bf, 512, 2048, 2048, 0);
  cvt(W_ctx, Wctx_bf, 1024, 1024, 1024, 0);

  k_build_xin<<<(B_ * D_ + 255) / 256, 256, 0, stream>>>(xt, mean_vit, state_h, x_in);

  // pre_att = pre_attn[0][:,2:578] @ W_fc^T + b_fc     (M=256,K=576,N=512)
  k_gemm_bf16<<<dim3(32, 4), 128, 0, stream>>>(pre_attn + 2, L_, Wfc_bf, b_fc, nullptr, preA, 512, 576, 0);
  // gates = [xt, mean_vit+h1, h0] @ [W_ih|W_hh]^T + b_ih + b_hh   (K=1536,N=2048)
  k_gemm_bf16<<<dim3(128, 4), 128, 0, stream>>>(x_in, 1536, Wcat_bf, b_ih, b_hh, gates, 2048, 1536, 0);
  // LSTM + h_attn + LN(q) -> xq[:,512:], ctx_in[:,512:], new_h0/new_c
  k_lstm_ln<<<32, 256, 0, stream>>>(gates, state_c, preA, ln_q_g, ln_q_b, out, ctx_in, xq);
  // qh = q @ W_q^T + b_q    (A = xq[:,512:], lda=1024)
  k_gemm_bf16<<<dim3(32, 4), 128, 0, stream>>>(xq + 512, 1024, Wq_bf, b_q, nullptr, qh, 512, 512, 0);
  // fused attention -> xq[:, :512], new_pre
  k_attention<<<B_, 256, 0, stream>>>(re_VIT, W_fc1, b_fc1, qh, pre_attn, out, xq);
  // xg = [x, q] @ W_glu^T + b_glu   (K=1024,N=1024)
  k_gemm_bf16<<<dim3(64, 4), 128, 0, stream>>>(xq, 1024, Wglu_bf, b_glu, nullptr, xg, 1024, 1024, 0);
  k_glu_ln<<<32, 256, 0, stream>>>(xg, ln_ff_g, ln_ff_b, glu_x, ln_x);
  // h_ff = relu(ln_x @ W_ff1^T + b_ff1)   (K=512,N=2048)
  k_gemm_bf16<<<dim3(128, 4), 128, 0, stream>>>(ln_x, 512, Wff1_bf, b_ff1, nullptr, h_ff, 2048, 512, 1);
  // ff = h_ff @ W_ff2^T + b_ff2           (K=2048,N=512)
  k_gemm_bf16<<<dim3(32, 4), 128, 0, stream>>>(h_ff, 2048, Wff2_bf, b_ff2, nullptr, ffb, 512, 2048, 0);
  k_addto_ctx<<<(B_ * 512 + 255) / 256, 256, 0, stream>>>(glu_x, ffb, ctx_in);
  // ctxg = [att_out, h_att] @ W_ctx^T + b_ctx   (K=1024,N=1024)
  k_gemm_bf16<<<dim3(64, 4), 128, 0, stream>>>(ctx_in, 1024, Wctx_bf, b_ctx, nullptr, ctxg, 1024, 1024, 0);
  k_final_glu<<<(B_ * 512 + 255) / 256, 256, 0, stream>>>(ctxg, out);
}
